// GNN_90348932038673
// MI455X (gfx1250) — matrix operand order repaired and checked by reading, not compiled, not run on hardware
//
#include <hip/hip_runtime.h>
#include <hip/hip_bf16.h>
#include <math.h>
#include <stdint.h>

#define N_APPS 20000
#define N_LOCS 5000
#define N_TIMES 168
#define N_NODES (N_APPS + N_LOCS + N_TIMES) /* 25168 = 16*1573 */
#define DIM 128
#define SEQ 64
#define BATCH 1024
#define NEDGE 500000

typedef __bf16 bf16_t;
typedef bf16_t v16bf __attribute__((ext_vector_type(16)));
typedef float  v8f   __attribute__((ext_vector_type(8)));

// ---------------------------------------------------------------- zero helpers
__global__ __launch_bounds__(256) void zero_f_kernel(float* __restrict__ p, int n) {
  int i = blockIdx.x * blockDim.x + threadIdx.x;
  int stride = gridDim.x * blockDim.x;
  for (; i < n; i += stride) p[i] = 0.0f;
}
__global__ __launch_bounds__(256) void zero_i_kernel(int* __restrict__ p, int n) {
  int i = blockIdx.x * blockDim.x + threadIdx.x;
  int stride = gridDim.x * blockDim.x;
  for (; i < n; i += stride) p[i] = 0;
}

// ---------------------------------------------------------------- sel_count scatter
__global__ __launch_bounds__(256) void count_kernel(const int* __restrict__ times,
                                                    const int* __restrict__ locs,
                                                    const int* __restrict__ apps,
                                                    float* __restrict__ sel) {
  int idx = blockIdx.x * blockDim.x + threadIdx.x;
  if (idx >= BATCH * (SEQ + 2)) return;
  int row = idx / (SEQ + 2);
  int p = idx - row * (SEQ + 2);
  int node;
  if (p == 0)      node = N_APPS + N_LOCS + times[row];
  else if (p == 1) node = N_APPS + locs[row];
  else             node = apps[row * SEQ + (p - 2)];
  atomicAdd(sel + node, 1.0f);
}

// ---------------------------------------------------------------- CSR build (once per launch)
__global__ __launch_bounds__(256) void deg_kernel(const int* __restrict__ edst,
                                                  int* __restrict__ deg) {
  int i = blockIdx.x * blockDim.x + threadIdx.x;
  int stride = gridDim.x * blockDim.x;
  for (; i < NEDGE; i += stride) atomicAdd(deg + edst[i], 1);
}

// single-workgroup chunked Hillis-Steele scan over 25168 degrees
__global__ __launch_bounds__(1024) void scan_kernel(const int* __restrict__ deg,
                                                    int* __restrict__ rowptr,
                                                    int* __restrict__ fill) {
  __shared__ int lds[1024];
  __shared__ int carry;
  if (threadIdx.x == 0) carry = 0;
  __syncthreads();
  for (int base = 0; base < N_NODES; base += 1024) {
    const int i = base + (int)threadIdx.x;
    const int v = (i < N_NODES) ? deg[i] : 0;
    lds[threadIdx.x] = v;
    __syncthreads();
    for (int off = 1; off < 1024; off <<= 1) {
      const int t = (threadIdx.x >= (unsigned)off) ? lds[threadIdx.x - off] : 0;
      __syncthreads();
      lds[threadIdx.x] += t;
      __syncthreads();
    }
    const int incl = lds[threadIdx.x] + carry;
    if (i < N_NODES) { rowptr[i + 1] = incl; fill[i] = incl - v; }
    __syncthreads();
    if (threadIdx.x == 1023) carry = incl;
    __syncthreads();
  }
  if (threadIdx.x == 0) rowptr[0] = 0;
}

__global__ __launch_bounds__(256) void scatter_kernel(const int* __restrict__ esrc,
                                                      const int* __restrict__ edst,
                                                      int* __restrict__ fill,
                                                      int* __restrict__ csr_src) {
  int i = blockIdx.x * blockDim.x + threadIdx.x;
  int stride = gridDim.x * blockDim.x;
  for (; i < NEDGE; i += stride) {
    const int pos = atomicAdd(fill + edst[i], 1);
    csr_src[pos] = esrc[i];
  }
}

// ---------------------------------------------------------------- weight pre-pack
// P[((kt*128 + col)*2 + half)*16 + e] = bf16(W[kt*32 + half*16 + e][col])
// -> each lane's 16-element B fragment is one contiguous, 32B-aligned v16bf.
__global__ __launch_bounds__(256) void pack_w_kernel(const float* __restrict__ W,
                                                     bf16_t* __restrict__ P) {
  const int idx = blockIdx.x * blockDim.x + threadIdx.x;
  if (idx >= 4 * DIM * 2 * 16) return;               // 16384
  const int e  = idx & 15;
  const int hf = (idx >> 4) & 1;
  const int c  = (idx >> 5) & 127;
  const int kt = idx >> 12;
  const int k  = kt * 32 + hf * 16 + e;
  P[idx] = (bf16_t)W[k * DIM + c];
}

// ---------------------------------------------------------------- atomic-free pull aggregation
// One wave32 per destination node; lane l owns dims [4l,4l+4). Writes the MEAN directly.
__global__ __launch_bounds__(256) void pull_mean_kernel(const float* __restrict__ H,
                                                        const float* __restrict__ sel,
                                                        const int* __restrict__ rowptr,
                                                        const int* __restrict__ csr_src,
                                                        float* __restrict__ mean) {
  const int lane = threadIdx.x & 31;
  const int node = (blockIdx.x * blockDim.x + threadIdx.x) >> 5;
  if (node >= N_NODES) return;
  const int beg = rowptr[node];
  const int end = rowptr[node + 1];
  float ax = 0.0f, ay = 0.0f, az = 0.0f, aw = 0.0f, cnt = 0.0f;
  for (int i = beg; i < end; ++i) {
    const int src = csr_src[i];
    const float w = sel[src];
    if (i + 1 < end) {  // hide L2 gather latency behind the FMAs
      __builtin_prefetch(H + (size_t)csr_src[i + 1] * DIM + lane * 4, 0, 1);
    }
    if (w == 0.0f) continue;                 // wave-uniform skip of dead edges
    const float4 hv = ((const float4*)(H + (size_t)src * DIM))[lane];
    ax += hv.x * w; ay += hv.y * w; az += hv.z * w; aw += hv.w * w;
    cnt += w;
  }
  const float inv = 1.0f / fmaxf(cnt, 1.0f);
  float4 m; m.x = ax * inv; m.y = ay * inv; m.z = az * inv; m.w = aw * inv;
  ((float4*)(mean + (size_t)node * DIM))[lane] = m;
}

// ---------------------------------------------------------------- WMMA layer
// out = relu(H @ Ws + MEAN @ Wn + bias); Ws/Wn pre-packed bf16 fragments.
__global__ __launch_bounds__(256) void gemm_layer_kernel(const float* __restrict__ H,
                                                         const float* __restrict__ MEAN,
                                                         const bf16_t* __restrict__ Pws,
                                                         const bf16_t* __restrict__ Pwn,
                                                         const float* __restrict__ bias,
                                                         float* __restrict__ out) {
  const int mtile = blockIdx.x;           // 0..1572
  const int ntile = threadIdx.x >> 5;     // wave id = column tile (0..7)
  const int lane  = threadIdx.x & 31;
  const int half  = lane >> 4;
  const int mr    = lane & 15;
  const int row   = mtile * 16 + mr;
  const int col   = ntile * 16 + mr;

  v8f acc = {};
#pragma unroll
  for (int kt = 0; kt < 4; ++kt) {
    v16bf ah, am;
    // A fragments in contiguous K order matching packed B: VGPR v: K=16*half+2v+{0,1}
#pragma unroll
    for (int v = 0; v < 8; ++v) {
      const int kb = kt * 32 + 2 * v + half * 16;
      const float2 hv = *(const float2*)(H    + (size_t)row * DIM + kb);
      const float2 mv = *(const float2*)(MEAN + (size_t)row * DIM + kb);
      ah[2 * v]     = (bf16_t)hv.x;
      ah[2 * v + 1] = (bf16_t)hv.y;
      am[2 * v]     = (bf16_t)mv.x;
      am[2 * v + 1] = (bf16_t)mv.y;
    }
    const int fidx = (kt * DIM + col) * 2 + half;    // 32B-aligned fragment index
    const v16bf bs = ((const v16bf*)Pws)[fidx];
    const v16bf bn = ((const v16bf*)Pwn)[fidx];
    acc = __builtin_amdgcn_wmma_f32_16x16x32_bf16(false, ah, false, bs, (short)0, acc, false, false);
    acc = __builtin_amdgcn_wmma_f32_16x16x32_bf16(false, am, false, bn, (short)0, acc, false, false);
  }
  const float bb = bias[col];
#pragma unroll
  for (int v = 0; v < 8; ++v) {
    const int orow = mtile * 16 + v + half * 8;      // C/D layout
    out[(size_t)orow * DIM + col] = fmaxf(acc[v] + bb, 0.0f);
  }
}

// ---------------------------------------------------------------- cosine-sim loss
__device__ __forceinline__ float wave_sum32(float x) {
#pragma unroll
  for (int m = 16; m >= 1; m >>= 1) x += __shfl_xor(x, m, 32);
  return x;
}
__device__ __forceinline__ float dot4(const float4 a, const float4 b) {
  return a.x * b.x + a.y * b.y + a.z * b.z + a.w * b.w;
}

__global__ __launch_bounds__(256) void sim_kernel(const float* __restrict__ H2,
                                                  const int* __restrict__ times,
                                                  const int* __restrict__ locs,
                                                  const int* __restrict__ apps,
                                                  float* __restrict__ loss) {
  const int lane = threadIdx.x & 31;
  const int b = (blockIdx.x * blockDim.x + threadIdx.x) >> 5;
  if (b >= BATCH) return;
  const int tnode = N_APPS + N_LOCS + times[b];
  const int lnode = N_APPS + locs[b];
  const float4 tv = ((const float4*)(H2 + (size_t)tnode * DIM))[lane];
  const float4 lv = ((const float4*)(H2 + (size_t)lnode * DIM))[lane];
  float4 wv;
  wv.x = 0.5f * (tv.x + lv.x); wv.y = 0.5f * (tv.y + lv.y);
  wv.z = 0.5f * (tv.z + lv.z); wv.w = 0.5f * (tv.w + lv.w);
  const float n_t = wave_sum32(dot4(tv, tv));
  const float n_l = wave_sum32(dot4(lv, lv));
  const float n_w = wave_sum32(dot4(wv, wv));

  float simt = 0.0f, siml = 0.0f, sima = 0.0f;
  for (int j = 0; j < SEQ; ++j) {
    const int an = apps[b * SEQ + j];
    const float4 av = ((const float4*)(H2 + (size_t)an * DIM))[lane];
    float4 u, v;
    u.x = 0.5f * (lv.x + av.x); u.y = 0.5f * (lv.y + av.y);
    u.z = 0.5f * (lv.z + av.z); u.w = 0.5f * (lv.w + av.w);
    v.x = 0.5f * (tv.x + av.x); v.y = 0.5f * (tv.y + av.y);
    v.z = 0.5f * (tv.z + av.z); v.w = 0.5f * (tv.w + av.w);
    const float s_tu = wave_sum32(dot4(tv, u));
    const float s_uu = wave_sum32(dot4(u, u));
    const float s_lv = wave_sum32(dot4(lv, v));
    const float s_vv = wave_sum32(dot4(v, v));
    const float s_aw = wave_sum32(dot4(av, wv));
    const float s_aa = wave_sum32(dot4(av, av));
    simt += s_tu / sqrtf(n_t * s_uu);
    siml += s_lv / sqrtf(n_l * s_vv);
    sima += s_aw / sqrtf(s_aa * n_w);
  }
  if (lane == 0) loss[b] = simt + siml + sima;
}

// ---------------------------------------------------------------- driver
extern "C" void kernel_launch(void* const* d_in, const int* in_sizes, int n_in,
                              void* d_out, int out_size, void* d_ws, size_t ws_size,
                              hipStream_t stream) {
  (void)in_sizes; (void)n_in; (void)out_size; (void)ws_size;
  const int*   times  = (const int*)d_in[1];
  const int*   locs   = (const int*)d_in[2];
  const int*   apps   = (const int*)d_in[3];
  const int*   ntimes = (const int*)d_in[4];
  const int*   nlocs  = (const int*)d_in[5];
  const int*   napps  = (const int*)d_in[6];
  const int*   esrc   = (const int*)d_in[7];
  const int*   edst   = (const int*)d_in[8];
  const float* emb    = (const float*)d_in[9];
  const float* Ws1    = (const float*)d_in[10];
  const float* Wn1    = (const float*)d_in[11];
  const float* b1     = (const float*)d_in[12];
  const float* Ws2    = (const float*)d_in[13];
  const float* Wn2    = (const float*)d_in[14];
  const float* b2     = (const float*)d_in[15];
  float*       outp   = (float*)d_out;

  // ---- workspace layout
  float* mean = (float*)d_ws;                                // NN*DIM
  float* h1   = mean + (size_t)N_NODES * DIM;                // NN*DIM
  float* h2   = h1   + (size_t)N_NODES * DIM;                // NN*DIM
  float* sel  = h2   + (size_t)N_NODES * DIM;                // NN
  int* deg     = (int*)(sel + N_NODES);                      // NN
  int* rowptr  = deg + N_NODES;                              // NN+1
  int* fill    = rowptr + N_NODES + 1;                       // NN
  int* csr_src = fill + N_NODES;                             // NEDGE
  uintptr_t pb = (uintptr_t)(csr_src + NEDGE);
  pb = (pb + 31) & ~(uintptr_t)31;                           // 32B align for v16bf loads
  bf16_t* P1s = (bf16_t*)pb;                                 // 16384 each
  bf16_t* P1n = P1s + 16384;
  bf16_t* P2s = P1n + 16384;
  bf16_t* P2n = P2s + 16384;

  const int CNT_BLOCKS  = (BATCH * (SEQ + 2) + 255) / 256;
  const int GEMM_BLOCKS = N_NODES / 16;                      // 1573, exact
  const int PULL_BLOCKS = (N_NODES + 7) / 8;                 // 8 waves/block, exact
  const int SIM_BLOCKS  = (BATCH + 7) / 8;

  // ---- once per launch: CSR by destination + packed weights
  zero_i_kernel<<<32, 256, 0, stream>>>(deg, N_NODES);
  deg_kernel<<<1024, 256, 0, stream>>>(edst, deg);
  scan_kernel<<<1, 1024, 0, stream>>>(deg, rowptr, fill);
  scatter_kernel<<<1024, 256, 0, stream>>>(esrc, edst, fill, csr_src);
  pack_w_kernel<<<64, 256, 0, stream>>>(Ws1, P1s);
  pack_w_kernel<<<64, 256, 0, stream>>>(Wn1, P1n);
  pack_w_kernel<<<64, 256, 0, stream>>>(Ws2, P2s);
  pack_w_kernel<<<64, 256, 0, stream>>>(Wn2, P2n);

  for (int pass = 0; pass < 2; ++pass) {
    const int* T = pass ? ntimes : times;
    const int* L = pass ? nlocs : locs;
    const int* A = pass ? napps : apps;
    float* lossp = outp + pass * BATCH;

    zero_f_kernel<<<32, 256, 0, stream>>>(sel, N_NODES);
    count_kernel<<<CNT_BLOCKS, 256, 0, stream>>>(T, L, A, sel);

    // layer 1: h1 = relu(emb@Ws1 + mean@Wn1 + b1)
    pull_mean_kernel<<<PULL_BLOCKS, 256, 0, stream>>>(emb, sel, rowptr, csr_src, mean);
    gemm_layer_kernel<<<GEMM_BLOCKS, 256, 0, stream>>>(emb, mean, P1s, P1n, b1, h1);

    // layer 2: h2 = relu(h1@Ws2 + mean@Wn2 + b2)
    pull_mean_kernel<<<PULL_BLOCKS, 256, 0, stream>>>(h1, sel, rowptr, csr_src, mean);
    gemm_layer_kernel<<<GEMM_BLOCKS, 256, 0, stream>>>(h1, mean, P2s, P2n, b2, h2);

    sim_kernel<<<SIM_BLOCKS, 256, 0, stream>>>(h2, T, L, A, lossp);
  }
}
